// PAM_74045236183531
// MI455X (gfx1250) — compile-verified, matrix operand order checked by reading, not run on hardware
//
#include <hip/hip_runtime.h>

// ---------------------------------------------------------------------------
// PAM (position attention) pipeline for gfx1250, bf16 WMMA everywhere.
// conv3x3+BN+ReLU -> 3x (1x1 proj+BN+ReLU) -> flash softmax(B^T C) -> D@attn
// Round 3: fix async-LDS builtin signature (takes int4 AS1/AS3 pointers).
// ---------------------------------------------------------------------------

#define DEV static __device__ __forceinline__
#define EPSX 1e-5f

typedef __attribute__((ext_vector_type(16))) __bf16         v16bf;
typedef __attribute__((ext_vector_type(16))) unsigned short v16u;
typedef __attribute__((ext_vector_type(8)))  unsigned short u16x8;
typedef __attribute__((ext_vector_type(8)))  float          v8f;
typedef __attribute__((ext_vector_type(4)))  int            v4i;

#if __has_builtin(__builtin_amdgcn_global_load_async_to_lds_b128)
#define HAS_ASYNC_LDS 1
#else
#define HAS_ASYNC_LDS 0
#endif

#define AS1 __attribute__((address_space(1)))
#define AS3 __attribute__((address_space(3)))

DEV void wait_async0() {
#if __has_builtin(__builtin_amdgcn_s_wait_asynccnt)
  __builtin_amdgcn_s_wait_asynccnt(0);
#else
  asm volatile("s_wait_asynccnt 0x0" ::: "memory");
#endif
}

#if HAS_ASYNC_LDS
// Copy 64 bytes (4 x b128) per lane from global to LDS, async (ASYNCcnt).
// INST_OFFSET applies to both global and LDS sides (ISA 15.18.1), so one
// immediate advances both addresses.
DEV void async_copy64(const unsigned short* g, unsigned short* l) {
  AS1 v4i* gp = (AS1 v4i*)(unsigned short*)g;
  AS3 v4i* lp = (AS3 v4i*)l;
  __builtin_amdgcn_global_load_async_to_lds_b128(gp, lp, 0, 0);
  __builtin_amdgcn_global_load_async_to_lds_b128(gp, lp, 16, 0);
  __builtin_amdgcn_global_load_async_to_lds_b128(gp, lp, 32, 0);
  __builtin_amdgcn_global_load_async_to_lds_b128(gp, lp, 48, 0);
}
#endif

DEV unsigned short f2bf(float f) {
  unsigned int u = __builtin_bit_cast(unsigned int, f);
  u += 0x7FFFu + ((u >> 16) & 1u);           // round-to-nearest-even
  return (unsigned short)(u >> 16);
}

DEV v8f vzero() { v8f z = {0.f,0.f,0.f,0.f,0.f,0.f,0.f,0.f}; return z; }

// Load a 16x32 bf16 WMMA operand fragment from a [16-dim-major][K-contig]
// source (global or LDS). Lane L: row = row0 + (L&15); per-lane K layout per
// ISA 7.12.2: elems 0-7 -> K = k0+8*half .. +7, elems 8-15 -> +16..+23.
DEV v16bf load_frag(const unsigned short* base, int row0, int stride, int k0, int lane) {
  const int mr = lane & 15, half = lane >> 4;
  const unsigned short* p = base + (size_t)(row0 + mr) * (size_t)stride + k0 + half * 8;
  u16x8 lo = *(const u16x8*)(p);
  u16x8 hi = *(const u16x8*)(p + 16);
  v16u v = __builtin_shufflevector(lo, hi, 0,1,2,3,4,5,6,7,8,9,10,11,12,13,14,15);
  return __builtin_bit_cast(v16bf, v);
}

DEV v8f wmma_bf16(v16bf a, v16bf b, v8f c) {
  return __builtin_amdgcn_wmma_f32_16x16x32_bf16(false, a, false, b, (short)0, c, false, false);
}

DEV float redmax16(float v) {
  #pragma unroll
  for (int m = 1; m < 16; m <<= 1) v = fmaxf(v, __shfl_xor(v, m, 16));
  return v;
}
DEV float redsum16(float v) {
  #pragma unroll
  for (int m = 1; m < 16; m <<= 1) v += __shfl_xor(v, m, 16);
  return v;
}

// ---------------------------------------------------------------------------
// Prep kernels: fold BN into weights/biases, convert to bf16.
// ---------------------------------------------------------------------------

__global__ __launch_bounds__(64) void k_prep_vec(
    const float* ba, const float* bb, const float* bc, const float* bd,
    const float* ga, const float* bea, const float* mea, const float* va,
    const float* gb, const float* beb, const float* meb, const float* vb,
    const float* gc, const float* bec, const float* mec, const float* vc,
    const float* gd, const float* bed, const float* med, const float* vd,
    float* dv) {
  int o = threadIdx.x;
  if (o < 64) {
    float g;
    g = ga[o] * rsqrtf(va[o] + EPSX); dv[o      ] = ba[o]*g + bea[o] - mea[o]*g;
    g = gb[o] * rsqrtf(vb[o] + EPSX); dv[64  + o] = bb[o]*g + beb[o] - meb[o]*g;
    g = gc[o] * rsqrtf(vc[o] + EPSX); dv[128 + o] = bc[o]*g + bec[o] - mec[o]*g;
    g = gd[o] * rsqrtf(vd[o] + EPSX); dv[192 + o] = bd[o]*g + bed[o] - med[o]*g;
  }
}

__global__ __launch_bounds__(256) void k_prep_x(const float* __restrict__ x,
                                                unsigned short* __restrict__ xb) {
  size_t i = (size_t)blockIdx.x * 256 + threadIdx.x;   // 4*512*64*64 elements
  xb[i] = f2bf(x[i]);
}

// Wa' [tap][o][c] bf16, BN gain folded in.  Wa is (64, 512, 3, 3).
__global__ __launch_bounds__(256) void k_prep_wa(const float* __restrict__ Wa,
                                                 const float* __restrict__ ga,
                                                 const float* __restrict__ va,
                                                 unsigned short* __restrict__ wab) {
  int idx = blockIdx.x * 256 + threadIdx.x;            // 9*64*512
  int tap = idx / (64 * 512);
  int rem = idx % (64 * 512);
  int o = rem / 512, c = rem % 512;
  int kh = tap / 3, kw = tap % 3;
  float g = ga[o] * rsqrtf(va[o] + EPSX);
  wab[idx] = f2bf(Wa[(((size_t)o * 512 + c) * 3 + kh) * 3 + kw] * g);
}

// W' [p][o][c] bf16 for the three 1x1 projections, BN gains folded.
__global__ __launch_bounds__(256) void k_prep_wp(const float* __restrict__ Wb,
                                                 const float* __restrict__ Wc,
                                                 const float* __restrict__ Wd,
                                                 const float* __restrict__ gb, const float* __restrict__ vb,
                                                 const float* __restrict__ gc, const float* __restrict__ vc,
                                                 const float* __restrict__ gd, const float* __restrict__ vd,
                                                 unsigned short* __restrict__ wp) {
  int idx = blockIdx.x * 256 + threadIdx.x;            // 3*64*64
  int p = idx / 4096, rem = idx % 4096;
  int o = rem / 64;
  const float* W = (p == 0) ? Wb : (p == 1) ? Wc : Wd;
  const float* g = (p == 0) ? gb : (p == 1) ? gc : gd;
  const float* v = (p == 0) ? vb : (p == 1) ? vc : vd;
  wp[idx] = f2bf(W[rem] * (g[o] * rsqrtf(v[o] + EPSX)));
}

// ---------------------------------------------------------------------------
// Conv 3x3 (512->64) as 9-tap GEMM. Block = one (batch, image row), 4 waves,
// wave w owns channel tile m0=16w x 64 pixels. x tile staged in LDS
// transposed to [pixel][channel] with zeroed halo.
// ---------------------------------------------------------------------------
__global__ __launch_bounds__(128) void k_conv(const unsigned short* __restrict__ xb,
                                              const unsigned short* __restrict__ wab,
                                              const float* __restrict__ dva,
                                              float* __restrict__ Af,
                                              unsigned short* __restrict__ At) {
  __shared__ __attribute__((aligned(16))) unsigned short xs[66 * 32]; // [p][c]
  const int h = blockIdx.x, b = blockIdx.y;
  const int tid = threadIdx.x, lane = tid & 31, wave = tid >> 5;
  const int m0 = wave * 16;
  const int half = lane >> 4, nc = lane & 15;

  v8f acc[4];
  #pragma unroll
  for (int nt = 0; nt < 4; ++nt) acc[nt] = vzero();

  for (int kh = 0; kh < 3; ++kh) {
    const int r = h + kh - 1;
    for (int c0 = 0; c0 < 512; c0 += 32) {
      __syncthreads();
      if (r < 0 || r > 63) {
        for (int e = tid; e < 66 * 32; e += 128) xs[e] = 0;
      } else {
        if (tid < 64) xs[(tid < 32 ? 0 : 65) * 32 + (tid & 31)] = 0;   // halo cols
        const unsigned short* xrow = xb + (((size_t)(b * 512 + c0) * 64 + r) * 64);
        for (int e = tid; e < 32 * 8; e += 128) {
          int c = e >> 3, chunk = e & 7;
          u16x8 v = *(const u16x8*)(xrow + (size_t)c * 4096 + chunk * 8);
          #pragma unroll
          for (int q = 0; q < 8; ++q) xs[(1 + chunk * 8 + q) * 32 + c] = v[q];
        }
      }
      __syncthreads();
      #pragma unroll
      for (int kw = 0; kw < 3; ++kw) {
        const int tap = kh * 3 + kw;
        v16bf a = load_frag(wab + (size_t)tap * 64 * 512, m0, 512, c0, lane);
        #pragma unroll
        for (int nt = 0; nt < 4; ++nt) {
          v16bf bf = load_frag(xs, nt * 16 + kw, 32, 0, lane);
          acc[nt] = wmma_bf16(a, bf, acc[nt]);
        }
      }
    }
  }

  float dr[8];
  #pragma unroll
  for (int i = 0; i < 8; ++i) dr[i] = dva[m0 + i + 8 * half];
  const size_t nimg = (size_t)h * 64;
  #pragma unroll
  for (int nt = 0; nt < 4; ++nt) {
    size_t n = nimg + nt * 16 + nc;
    #pragma unroll
    for (int i = 0; i < 8; ++i) {
      int o = m0 + i + 8 * half;
      float v = acc[nt][i] + dr[i];
      v = v > 0.f ? v : 0.f;
      Af[((size_t)b * 64 + o) * 4096 + n] = v;          // A  [b][cls][n] fp32
      At[((size_t)b * 4096 + n) * 64 + o] = f2bf(v);    // A^T [b][n][cls] bf16
    }
  }
}

// ---------------------------------------------------------------------------
// 1x1 projections: B^T, C^T stored [n][cls]; D stored [cls][n].
// ---------------------------------------------------------------------------
__global__ __launch_bounds__(128) void k_proj(const unsigned short* __restrict__ At,
                                              const unsigned short* __restrict__ wp,
                                              const float* __restrict__ dvb,
                                              unsigned short* __restrict__ Bt,
                                              unsigned short* __restrict__ Ct,
                                              unsigned short* __restrict__ Dp) {
  const int mblk = blockIdx.x, b = blockIdx.y;
  const int tid = threadIdx.x, lane = tid & 31, wave = tid >> 5;
  const int half = lane >> 4, nc = lane & 15;
  const int m0 = mblk * 64 + wave * 16;
  const unsigned short* arow = At + (size_t)b * 4096 * 64;
  v16bf a0 = load_frag(arow, m0, 64, 0, lane);
  v16bf a1 = load_frag(arow, m0, 64, 32, lane);

  for (int p = 0; p < 3; ++p) {
    const unsigned short* w = wp + (size_t)p * 64 * 64;
    const float* d = dvb + p * 64;
    #pragma unroll
    for (int ot = 0; ot < 4; ++ot) {
      v8f acc = vzero();
      acc = wmma_bf16(a0, load_frag(w, ot * 16, 64, 0, lane), acc);
      acc = wmma_bf16(a1, load_frag(w, ot * 16, 64, 32, lane), acc);
      const int o = ot * 16 + nc;
      const float bias = d[o];
      #pragma unroll
      for (int i = 0; i < 8; ++i) {
        int m = m0 + i + 8 * half;
        float v = acc[i] + bias;
        v = v > 0.f ? v : 0.f;
        unsigned short bf = f2bf(v);
        if (p == 0)      Bt[((size_t)b * 4096 + m) * 64 + o] = bf;
        else if (p == 1) Ct[((size_t)b * 4096 + m) * 64 + o] = bf;
        else             Dp[((size_t)b * 64 + o) * 4096 + m] = bf;
      }
    }
  }
}

// ---------------------------------------------------------------------------
// Flash attention: per 64-query block, stream 64-key blocks via LDS
// (async global->LDS DMA), online softmax, O = P@V accumulated fp32.
// out = A + alpha * O/rowsum.
// ---------------------------------------------------------------------------
__global__ __launch_bounds__(128) void k_attn(const unsigned short* __restrict__ Bt,
                                              const unsigned short* __restrict__ Ct,
                                              const unsigned short* __restrict__ Dp,
                                              const float* __restrict__ Af,
                                              const float* __restrict__ alpha_p,
                                              float* __restrict__ out) {
  __shared__ __attribute__((aligned(16))) unsigned short Ks[64 * 64];      // [j][d]
  __shared__ __attribute__((aligned(16))) unsigned short Vs[64 * 64];      // [c][j]
  __shared__ __attribute__((aligned(16))) unsigned short Ps[4][16 * 64];   // per-wave [m][j]
  const int mblk = blockIdx.x, b = blockIdx.y;
  const int tid = threadIdx.x, lane = tid & 31, wave = tid >> 5;
  const int half = lane >> 4, nc = lane & 15;
  const int m0 = mblk * 64 + wave * 16;

  const unsigned short* Qg = Bt + (size_t)b * 4096 * 64;
  const unsigned short* Kg = Ct + (size_t)b * 4096 * 64;
  const unsigned short* Vg = Dp + (size_t)b * 64 * 4096;

  v16bf q0 = load_frag(Qg, m0, 64, 0, lane);
  v16bf q1 = load_frag(Qg, m0, 64, 32, lane);

  v8f O[4];
  float rmax[8], rsum[8];
  #pragma unroll
  for (int ct = 0; ct < 4; ++ct) O[ct] = vzero();
  #pragma unroll
  for (int i = 0; i < 8; ++i) { rmax[i] = -3.0e38f; rsum[i] = 0.f; }

  unsigned short* pw = Ps[wave];
  const int lrow = tid >> 1;               // 0..63: tile row owned by this thread
  const int lcol = (tid & 1) * 32;         // two 64-byte half-rows per row

  for (int j0 = 0; j0 < 4096; j0 += 64) {
#if HAS_ASYNC_LDS
    // Async DMA the K/V tiles straight into LDS (ASYNCcnt-tracked).
    async_copy64(Kg + (size_t)(j0 + lrow) * 64 + lcol, Ks + lrow * 64 + lcol);
    async_copy64(Vg + (size_t)lrow * 4096 + j0 + lcol, Vs + lrow * 64 + lcol);
    wait_async0();
#else
    for (int e = tid; e < 64 * 8; e += 128) {
      int row = e >> 3, ch = e & 7;
      *(u16x8*)(Ks + row * 64 + ch * 8) = *(const u16x8*)(Kg + (size_t)(j0 + row) * 64 + ch * 8);
      *(u16x8*)(Vs + row * 64 + ch * 8) = *(const u16x8*)(Vg + (size_t)row * 4096 + j0 + ch * 8);
    }
#endif
    __syncthreads();

    if (j0 + 64 < 4096) {                   // prefetch next tiles into caches
      __builtin_prefetch(Kg + (size_t)(j0 + 64 + lrow) * 64 + lcol, 0, 1);
      __builtin_prefetch(Vg + (size_t)lrow * 4096 + j0 + 64 + lcol, 0, 1);
    }

    v8f lt[4];                              // logits [16 m][64 j]
    #pragma unroll
    for (int jt = 0; jt < 4; ++jt) {
      v8f z = vzero();
      z = wmma_bf16(q0, load_frag(Ks, jt * 16, 64, 0, lane), z);
      z = wmma_bf16(q1, load_frag(Ks, jt * 16, 64, 32, lane), z);
      lt[jt] = z;
    }

    #pragma unroll
    for (int i = 0; i < 8; ++i) {           // online softmax per row
      float mx = fmaxf(fmaxf(lt[0][i], lt[1][i]), fmaxf(lt[2][i], lt[3][i]));
      mx = redmax16(mx);
      float nm = fmaxf(rmax[i], mx);
      float f = __expf(rmax[i] - nm);
      rmax[i] = nm;
      float s = 0.f;
      #pragma unroll
      for (int jt = 0; jt < 4; ++jt) {
        float pv = __expf(lt[jt][i] - nm);
        lt[jt][i] = pv;
        s += pv;
      }
      s = redsum16(s);
      rsum[i] = rsum[i] * f + s;
      #pragma unroll
      for (int ct = 0; ct < 4; ++ct) O[ct][i] *= f;
    }

    #pragma unroll
    for (int jt = 0; jt < 4; ++jt)          // P -> LDS (C-layout -> A-layout)
      #pragma unroll
      for (int i = 0; i < 8; ++i)
        pw[(i + 8 * half) * 64 + jt * 16 + nc] = f2bf(lt[jt][i]);

    v16bf p0 = load_frag(pw, 0, 64, 0, lane);
    v16bf p1 = load_frag(pw, 0, 64, 32, lane);
    #pragma unroll
    for (int ct = 0; ct < 4; ++ct) {
      O[ct] = wmma_bf16(p0, load_frag(Vs, ct * 16, 64, 0, lane), O[ct]);
      O[ct] = wmma_bf16(p1, load_frag(Vs, ct * 16, 64, 32, lane), O[ct]);
    }
    __syncthreads();
  }

  const float al = alpha_p[0];
  #pragma unroll
  for (int ct = 0; ct < 4; ++ct) {
    int c = ct * 16 + nc;
    #pragma unroll
    for (int i = 0; i < 8; ++i) {
      int m = m0 + i + 8 * half;
      size_t idx = ((size_t)b * 64 + c) * 4096 + m;
      out[idx] = Af[idx] + al * (O[ct][i] / rsum[i]);
    }
  }
}

// ---------------------------------------------------------------------------
// Workspace layout (bytes), all region sizes multiples of 256.
// ---------------------------------------------------------------------------
static constexpr size_t OFF_XB = 0;                                        // x bf16
static constexpr size_t OFF_WA = OFF_XB + (size_t)4 * 512 * 4096 * 2;      // Wa' bf16
static constexpr size_t OFF_WP = OFF_WA + (size_t)9 * 64 * 512 * 2;        // W' bf16
static constexpr size_t OFF_DV = OFF_WP + (size_t)3 * 64 * 64 * 2;         // fused biases
static constexpr size_t OFF_AF = OFF_DV + (size_t)256 * 4;                 // A fp32
static constexpr size_t OFF_AT = OFF_AF + (size_t)4 * 64 * 4096 * 4;       // A^T bf16
static constexpr size_t OFF_BT = OFF_AT + (size_t)4 * 4096 * 64 * 2;       // B^T bf16
static constexpr size_t OFF_CT = OFF_BT + (size_t)4 * 4096 * 64 * 2;       // C^T bf16
static constexpr size_t OFF_DP = OFF_CT + (size_t)4 * 4096 * 64 * 2;       // D bf16

extern "C" void kernel_launch(void* const* d_in, const int* in_sizes, int n_in,
                              void* d_out, int out_size, void* d_ws, size_t ws_size,
                              hipStream_t stream) {
  (void)in_sizes; (void)n_in; (void)out_size; (void)ws_size;
  const float* x   = (const float*)d_in[0];
  const float* Wa  = (const float*)d_in[1];
  const float* ba  = (const float*)d_in[2];
  const float* Wb  = (const float*)d_in[3];
  const float* bb  = (const float*)d_in[4];
  const float* Wc  = (const float*)d_in[5];
  const float* bc  = (const float*)d_in[6];
  const float* Wd  = (const float*)d_in[7];
  const float* bd  = (const float*)d_in[8];
  const float* g_a = (const float*)d_in[9],  *be_a = (const float*)d_in[10],
             * m_a = (const float*)d_in[11], *v_a  = (const float*)d_in[12];
  const float* g_b = (const float*)d_in[13], *be_b = (const float*)d_in[14],
             * m_b = (const float*)d_in[15], *v_b  = (const float*)d_in[16];
  const float* g_c = (const float*)d_in[17], *be_c = (const float*)d_in[18],
             * m_c = (const float*)d_in[19], *v_c  = (const float*)d_in[20];
  const float* g_d = (const float*)d_in[21], *be_d = (const float*)d_in[22],
             * m_d = (const float*)d_in[23], *v_d  = (const float*)d_in[24];
  const float* alpha = (const float*)d_in[25];

  char* ws = (char*)d_ws;
  unsigned short* xb  = (unsigned short*)(ws + OFF_XB);
  unsigned short* wab = (unsigned short*)(ws + OFF_WA);
  unsigned short* wp  = (unsigned short*)(ws + OFF_WP);
  float*          dv  = (float*)(ws + OFF_DV);
  float*          Af  = (float*)(ws + OFF_AF);
  unsigned short* At  = (unsigned short*)(ws + OFF_AT);
  unsigned short* Bt  = (unsigned short*)(ws + OFF_BT);
  unsigned short* Ct  = (unsigned short*)(ws + OFF_CT);
  unsigned short* Dp  = (unsigned short*)(ws + OFF_DP);

  k_prep_vec<<<1, 64, 0, stream>>>(ba, bb, bc, bd,
                                   g_a, be_a, m_a, v_a,
                                   g_b, be_b, m_b, v_b,
                                   g_c, be_c, m_c, v_c,
                                   g_d, be_d, m_d, v_d, dv);
  k_prep_x <<<32768, 256, 0, stream>>>(x, xb);
  k_prep_wa<<<1152,  256, 0, stream>>>(Wa, g_a, v_a, wab);
  k_prep_wp<<<48,    256, 0, stream>>>(Wb, Wc, Wd, g_b, v_b, g_c, v_c, g_d, v_d, wp);

  dim3 grid(64, 4);
  k_conv<<<grid, 128, 0, stream>>>(xb, wab, dv, Af, At);
  k_proj<<<grid, 128, 0, stream>>>(At, wp, dv + 64, Bt, Ct, Dp);
  k_attn<<<grid, 128, 0, stream>>>(Bt, Ct, Dp, Af, alpha, (float*)d_out);
}
